// TPACausalSelfAttention_3590592660049
// MI455X (gfx1250) — matrix operand order repaired and checked by reading, not compiled, not run on hardware
//
#include <hip/hip_runtime.h>
#include <hip/hip_bf16.h>
#include <stdint.h>

// ---------------------------------------------------------------------------
// Problem constants (TPA causal self-attention)
// ---------------------------------------------------------------------------
#define BB 2
#define TT 2048
#define EE 2048
#define HH 16
#define DD 128
#define RQ 6
#define RK 2
#define RV 2
#define BT (BB * TT)      // 4096 tokens
#define NPROJ 1536        // padded concat width (1440 real + 96 zero pad)

// column offsets inside the concatenated projection
#define OFF_AQ 0          // RQ*H = 96
#define OFF_BQ 96         // RQ*D = 768
#define OFF_AK 864        // RK*H = 32
#define OFF_BK 896        // RK*D = 256
#define OFF_AV 1152       // RV*H = 32
#define OFF_BV 1184       // RV*D = 256
#define NREAL 1440

typedef __bf16 bf16;
typedef bf16 bf16x8 __attribute__((ext_vector_type(8)));
typedef bf16 bf16x16 __attribute__((ext_vector_type(16)));
typedef float f32x8 __attribute__((ext_vector_type(8)));
typedef int i32x4 __attribute__((ext_vector_type(4)));

// ---------------------------------------------------------------------------
// CDNA5 async global->LDS path (ASYNCcnt-tracked DMA-style copies).
// Signature (confirmed by compiler diagnostics):
//   (int4 addrspace(1)* gsrc, int4 addrspace(3)* lds_dst, imm offset, imm cpol)
// ---------------------------------------------------------------------------
#if defined(__has_builtin)
#if __has_builtin(__builtin_amdgcn_global_load_async_to_lds_b128)
#define USE_ASYNC_LDS 1
#endif
#endif

#ifdef USE_ASYNC_LDS
typedef __attribute__((address_space(1))) i32x4* g_i32x4_ptr;
typedef __attribute__((address_space(3))) i32x4* l_i32x4_ptr;
// Generic LDS pointers carry the LDS byte offset in addr[31:0] (flat
// aperture rule), so the uintptr round-trip yields a valid as(3) pointer.
__device__ __forceinline__ void async_copy_b128(void* lds_dst, const void* gsrc) {
  __builtin_amdgcn_global_load_async_to_lds_b128(
      (g_i32x4_ptr)(uintptr_t)gsrc, (l_i32x4_ptr)(uint32_t)(uintptr_t)lds_dst,
      /*offset=*/0, /*cpol=*/0);
}
#endif

// ---------------------------------------------------------------------------
// WMMA helpers (CDNA5 16x16x32 bf16, fp32 accumulate)
// ---------------------------------------------------------------------------
__device__ __forceinline__ f32x8 wmma_bf16(bf16x16 a, bf16x16 b, f32x8 c) {
  return __builtin_amdgcn_wmma_f32_16x16x32_bf16(
      /*neg_a=*/false, a, /*neg_b=*/false, b,
      /*c_mod=*/(short)0, c, /*reuse_a=*/false, /*reuse_b=*/false);
}

// A-matrix fragment (16x32, M x K), source row-major with leading dim `ld`.
// ISA 7.12.2: lanes 0-15 hold M=lane, K in {0..7, 16..23}; lanes 16-31 hold
// K in {8..15, 24..31}; 2 packed bf16 per VGPR.
__device__ __forceinline__ bf16x16 ld_frag_a(const bf16* tile, int ld) {
  const int l = threadIdx.x & 31;
  const bf16* p = tile + (size_t)(l & 15) * ld + ((l < 16) ? 0 : 8);
  bf16x8 lo = *(const bf16x8*)p;
  bf16x8 hi = *(const bf16x8*)(p + 16);
  bf16x16 r;
#pragma unroll
  for (int i = 0; i < 8; ++i) { r[i] = lo[i]; r[i + 8] = hi[i]; }
  return r;
}

// B-matrix fragment (32x16, K x N) supplied as B^T in memory: N x K row-major
// with leading dim `ld`. Lanes 0-15 hold K in {0..7, 16..23}; lanes 16-31 hold K in {8..15, 24..31}.
__device__ __forceinline__ bf16x16 ld_frag_b(const bf16* tileT, int ld) {
  const int l = threadIdx.x & 31;
  const bf16* p = tileT + (size_t)(l & 15) * ld + ((l < 16) ? 0 : 8);
  bf16x8 lo = *(const bf16x8*)p;
  bf16x8 hi = *(const bf16x8*)(p + 16);
  bf16x16 r;
#pragma unroll
  for (int i = 0; i < 8; ++i) { r[i] = lo[i]; r[i + 8] = hi[i]; }
  return r;
}

// ---------------------------------------------------------------------------
// fp32 -> bf16 conversion (grid-stride)
// ---------------------------------------------------------------------------
__global__ void __launch_bounds__(256) f2bf_kernel(const float* __restrict__ s,
                                                   bf16* __restrict__ d, long n) {
  long i = (long)blockIdx.x * blockDim.x + threadIdx.x;
  const long stride = (long)gridDim.x * blockDim.x;
  for (; i < n; i += stride) d[i] = (bf16)s[i];
}

// Build concatenated N x K weight block [aQ | bQ | aK | bK | aV | bV | 0-pad]
__global__ void __launch_bounds__(256) build_wcat_kernel(
    const float* __restrict__ aQ, const float* __restrict__ bQ,
    const float* __restrict__ aK, const float* __restrict__ bK,
    const float* __restrict__ aV, const float* __restrict__ bV,
    bf16* __restrict__ w) {
  const long i = (long)blockIdx.x * blockDim.x + threadIdx.x;
  if (i >= (long)NPROJ * EE) return;
  const int rowN = (int)(i >> 11);       // / EE
  const int col = (int)(i & (EE - 1));
  float v = 0.f;
  if (rowN < OFF_BQ)            v = aQ[(rowN - OFF_AQ) * EE + col];
  else if (rowN < OFF_AK)       v = bQ[(rowN - OFF_BQ) * EE + col];
  else if (rowN < OFF_BK)       v = aK[(rowN - OFF_AK) * EE + col];
  else if (rowN < OFF_AV)       v = bK[(rowN - OFF_BK) * EE + col];
  else if (rowN < OFF_BV)       v = aV[(rowN - OFF_AV) * EE + col];
  else if (rowN < NREAL)        v = bV[(rowN - OFF_BV) * EE + col];
  w[i] = (bf16)v;
}

// ---------------------------------------------------------------------------
// Generic bf16 WMMA GEMM:  C[M x N] = A[M x K] (row-major) * B^T[N x K]
// Block = 256 threads (8 waves) arranged 4(M) x 2(N) -> 128x64 C tile;
// each wave owns a 32x32 sub-tile (4 WMMA per 32-wide K-step).
// K loop is ping-pong unrolled by 64 with two independent fragment sets so
// there are no cross-iteration register copies: each half-step's
// global_load_b128s fly under the other half-step's 4 v_wmma issues.
// Requires K % 64 == 0 (true for all launches here: K = 2048).
// ---------------------------------------------------------------------------
__global__ void __launch_bounds__(256) wmma_gemm_bf16(
    const bf16* __restrict__ A, const bf16* __restrict__ Bt,
    float* __restrict__ C, int K, int lda, int ldb, int ldc) {
  const int wave = threadIdx.x >> 5;
  const int lane = threadIdx.x & 31;
  const int m0 = blockIdx.y * 128 + (wave >> 1) * 32;
  const int n0 = blockIdx.x * 64 + (wave & 1) * 32;

  const bf16* Ap0 = A + (size_t)m0 * lda;
  const bf16* Ap1 = A + (size_t)(m0 + 16) * lda;
  const bf16* Bp0 = Bt + (size_t)n0 * ldb;
  const bf16* Bp1 = Bt + (size_t)(n0 + 16) * ldb;

  f32x8 acc00, acc01, acc10, acc11;
#pragma unroll
  for (int i = 0; i < 8; ++i) {
    acc00[i] = 0.f; acc01[i] = 0.f; acc10[i] = 0.f; acc11[i] = 0.f;
  }

  // prologue: fragment set 0 at k = 0
  bf16x16 a0 = ld_frag_a(Ap0, lda), a1 = ld_frag_a(Ap1, lda);
  bf16x16 b0 = ld_frag_b(Bp0, ldb), b1 = ld_frag_b(Bp1, ldb);

  for (int k = 0; k < K; k += 64) {
    const int k1 = k + 32;               // always < K (K % 64 == 0)
    if (k1 + 480 < K) {
      __builtin_prefetch(Ap0 + (size_t)(lane & 15) * lda + k1 + 480, 0, 1);
      __builtin_prefetch(Bp0 + (size_t)(lane & 15) * ldb + k1 + 480, 0, 1);
    }
    // fragment set 1 at k + 32 (independent registers, no rotation copies)
    const bf16x16 c0 = ld_frag_a(Ap0 + k1, lda);
    const bf16x16 c1 = ld_frag_a(Ap1 + k1, lda);
    const bf16x16 d0 = ld_frag_b(Bp0 + k1, ldb);
    const bf16x16 d1 = ld_frag_b(Bp1 + k1, ldb);

    acc00 = wmma_bf16(a0, b0, acc00);
    acc01 = wmma_bf16(a0, b1, acc01);
    acc10 = wmma_bf16(a1, b0, acc10);
    acc11 = wmma_bf16(a1, b1, acc11);

    const int k2 = k + 64;
    if (k2 < K) {                        // reload set 0 for next iteration
      a0 = ld_frag_a(Ap0 + k2, lda);
      a1 = ld_frag_a(Ap1 + k2, lda);
      b0 = ld_frag_b(Bp0 + k2, ldb);
      b1 = ld_frag_b(Bp1 + k2, ldb);
    }
    acc00 = wmma_bf16(c0, d0, acc00);
    acc01 = wmma_bf16(c0, d1, acc01);
    acc10 = wmma_bf16(c1, d0, acc10);
    acc11 = wmma_bf16(c1, d1, acc11);
  }

#pragma unroll
  for (int r = 0; r < 8; ++r) {
    const int mA = m0 + ((lane < 16) ? r : r + 8);
    const int mB = mA + 16;
    const int nA = n0 + (lane & 15);
    C[(size_t)mA * ldc + nA]      = acc00[r];
    C[(size_t)mA * ldc + nA + 16] = acc01[r];
    C[(size_t)mB * ldc + nA]      = acc10[r];
    C[(size_t)mB * ldc + nA + 16] = acc11[r];
  }
}

// ---------------------------------------------------------------------------
// TPA rank combine + RoPE + rmsnorm.
// One wave per (token, head); lane owns d in {lane, lane+32, lane+64, lane+96}.
// Writes Q (scaled by 1/sqrt(D)), K as (B,H,T,D) bf16; V transposed (B,H,D,T).
// ---------------------------------------------------------------------------
__global__ void __launch_bounds__(256) tpa_combine_kernel(
    const float* __restrict__ proj, const float* __restrict__ cosb,
    const float* __restrict__ sinb, bf16* __restrict__ qb,
    bf16* __restrict__ kbuf, bf16* __restrict__ vtb) {
  const int wave = threadIdx.x >> 5;
  const int lane = threadIdx.x & 31;
  const long idx = (long)blockIdx.x * 8 + wave;  // (b*T + t)*H + h
  const int h = (int)(idx & (HH - 1));
  const long bt = idx >> 4;
  const int t = (int)(bt & (TT - 1));
  const int b = (int)(bt >> 11);

  const float* pr = proj + bt * NPROJ;
  const float ca = cosb[t * 64 + lane], cb = cosb[t * 64 + 32 + lane];
  const float sa = sinb[t * 64 + lane], sb = sinb[t * 64 + 32 + lane];
  const float eps = 1.1920929e-07f;

  // ---- Q: rank-6 combine with RoPE on b, then rmsnorm, fold 1/sqrt(D) ----
  {
    float q0 = 0.f, q1 = 0.f, q2 = 0.f, q3 = 0.f;
#pragma unroll
    for (int r = 0; r < RQ; ++r) {
      const float a = pr[OFF_AQ + r * HH + h];
      const float* bp = pr + OFF_BQ + r * DD;
      const float x1a = bp[lane], x1b = bp[32 + lane];
      const float x2a = bp[64 + lane], x2b = bp[96 + lane];
      q0 += a * (x1a * ca + x2a * sa);
      q1 += a * (x1b * cb + x2b * sb);
      q2 += a * (-x1a * sa + x2a * ca);
      q3 += a * (-x1b * sb + x2b * cb);
    }
    const float inv = 1.f / RQ;
    q0 *= inv; q1 *= inv; q2 *= inv; q3 *= inv;
    float ss = q0 * q0 + q1 * q1 + q2 * q2 + q3 * q3;
#pragma unroll
    for (int m = 16; m >= 1; m >>= 1) ss += __shfl_xor(ss, m, 32);
    const float sc = rsqrtf(ss * (1.f / DD) + eps) * 0.08838834764831845f; // *1/sqrt(128)
    bf16* qo = qb + (((long)b * HH + h) * TT + t) * DD;
    qo[lane] = (bf16)(q0 * sc); qo[32 + lane] = (bf16)(q1 * sc);
    qo[64 + lane] = (bf16)(q2 * sc); qo[96 + lane] = (bf16)(q3 * sc);
  }

  // ---- K: rank-2 combine with RoPE, rmsnorm ----
  {
    float k0 = 0.f, k1 = 0.f, k2 = 0.f, k3 = 0.f;
#pragma unroll
    for (int r = 0; r < RK; ++r) {
      const float a = pr[OFF_AK + r * HH + h];
      const float* bp = pr + OFF_BK + r * DD;
      const float x1a = bp[lane], x1b = bp[32 + lane];
      const float x2a = bp[64 + lane], x2b = bp[96 + lane];
      k0 += a * (x1a * ca + x2a * sa);
      k1 += a * (x1b * cb + x2b * sb);
      k2 += a * (-x1a * sa + x2a * ca);
      k3 += a * (-x1b * sb + x2b * cb);
    }
    const float inv = 1.f / RK;
    k0 *= inv; k1 *= inv; k2 *= inv; k3 *= inv;
    float ss = k0 * k0 + k1 * k1 + k2 * k2 + k3 * k3;
#pragma unroll
    for (int m = 16; m >= 1; m >>= 1) ss += __shfl_xor(ss, m, 32);
    const float sc = rsqrtf(ss * (1.f / DD) + eps);
    bf16* ko = kbuf + (((long)b * HH + h) * TT + t) * DD;
    ko[lane] = (bf16)(k0 * sc); ko[32 + lane] = (bf16)(k1 * sc);
    ko[64 + lane] = (bf16)(k2 * sc); ko[96 + lane] = (bf16)(k3 * sc);
  }

  // ---- V: rank-2 combine, no rope / no norm, stored transposed (D x T) ----
  {
    float v0 = 0.f, v1 = 0.f, v2 = 0.f, v3 = 0.f;
#pragma unroll
    for (int r = 0; r < RV; ++r) {
      const float a = pr[OFF_AV + r * HH + h];
      const float* bp = pr + OFF_BV + r * DD;
      v0 += a * bp[lane];      v1 += a * bp[32 + lane];
      v2 += a * bp[64 + lane]; v3 += a * bp[96 + lane];
    }
    const float inv = 1.f / RV;
    bf16* vo = vtb + ((long)b * HH + h) * DD * TT + t;
    vo[(long)lane * TT] = (bf16)(v0 * inv);
    vo[(long)(32 + lane) * TT] = (bf16)(v1 * inv);
    vo[(long)(64 + lane) * TT] = (bf16)(v2 * inv);
    vo[(long)(96 + lane) * TT] = (bf16)(v3 * inv);
  }
}

// ---------------------------------------------------------------------------
// Flash-style causal attention, all-WMMA.
// Block = 128 threads (4 waves) owns a 64-row Q tile of one (b,h).
// K tile (64xD) and V^T tile (Dx64) staged in LDS via the CDNA5 async
// global->LDS path; Q frags stay VGPR-resident.
// ---------------------------------------------------------------------------
__global__ void __launch_bounds__(128) attn_kernel(
    const bf16* __restrict__ qb, const bf16* __restrict__ kbuf,
    const bf16* __restrict__ vtb, bf16* __restrict__ yb) {
  __shared__ bf16 ks[64][136];     // K tile:   kv-row  x D   (padded stride)
  __shared__ bf16 vs[128][72];     // V^T tile: d-row   x kv
  __shared__ bf16 ps[4][16][72];   // per-wave P transpose staging

  const int tid = threadIdx.x;
  const int wave = tid >> 5;
  const int lane = tid & 31;
  const int m0 = blockIdx.x * 64;
  const int h = blockIdx.y;
  const int b = blockIdx.z;
  const size_t headoff = ((size_t)b * HH + h) * (size_t)TT * DD;

  // Q fragments for this wave's 16 rows, all of D, VGPR-resident.
  bf16x16 qa[4];
  const bf16* qrow = qb + headoff + (size_t)(m0 + wave * 16) * DD;
#pragma unroll
  for (int kst = 0; kst < 4; ++kst) qa[kst] = ld_frag_a(qrow + kst * 32, DD);

  f32x8 o[8];
  float rm[8], rl[8];
#pragma unroll
  for (int i = 0; i < 8; ++i) {
    rm[i] = -3.0e38f; rl[i] = 0.f;
#pragma unroll
    for (int e = 0; e < 8; ++e) o[i][e] = 0.f;
  }

  const int jmax = m0 >> 6;
  for (int j = 0; j <= jmax; ++j) {
    // ---- stage K and V^T tiles into LDS ----
    {
      const bf16* kt = kbuf + headoff + (size_t)j * 64 * DD;
      const bf16* vt = vtb + headoff + j * 64;      // (d, t) layout
#ifdef USE_ASYNC_LDS
#pragma unroll
      for (int i = 0; i < 8; ++i) {
        const int idx = tid + i * 128;              // 0..1023
        const int row = idx >> 4, c = (idx & 15) * 8;
        async_copy_b128(&ks[row][c], kt + (size_t)row * DD + c);
      }
#pragma unroll
      for (int i = 0; i < 8; ++i) {
        const int idx = tid + i * 128;
        const int row = idx >> 3, c = (idx & 7) * 8;
        async_copy_b128(&vs[row][c], vt + (size_t)row * TT + c);
      }
      asm volatile("s_wait_asynccnt 0" ::: "memory");
#else
#pragma unroll
      for (int i = 0; i < 8; ++i) {
        const int idx = tid + i * 128;              // 0..1023
        const int row = idx >> 4, c = (idx & 15) * 8;
        *(bf16x8*)&ks[row][c] = *(const bf16x8*)(kt + (size_t)row * DD + c);
      }
#pragma unroll
      for (int i = 0; i < 8; ++i) {
        const int idx = tid + i * 128;
        const int row = idx >> 3, c = (idx & 7) * 8;
        *(bf16x8*)&vs[row][c] = *(const bf16x8*)(vt + (size_t)row * TT + c);
      }
#endif
    }
    __syncthreads();

    // ---- S = Q * K^T  (16 x 64 per wave) ----
    f32x8 s[4];
#pragma unroll
    for (int nf = 0; nf < 4; ++nf)
#pragma unroll
      for (int e = 0; e < 8; ++e) s[nf][e] = 0.f;
#pragma unroll
    for (int kst = 0; kst < 4; ++kst) {
      const bf16x16 a = qa[kst];
#pragma unroll
      for (int nf = 0; nf < 4; ++nf) {
        const bf16x16 bk = ld_frag_b(&ks[nf * 16][kst * 32], 136);
        s[nf] = wmma_bf16(a, bk, s[nf]);
      }
    }

    // ---- causal mask (diagonal tile only) ----
    if (j == jmax) {
#pragma unroll
      for (int nf = 0; nf < 4; ++nf) {
        const int ng = j * 64 + nf * 16 + (lane & 15);
#pragma unroll
        for (int r = 0; r < 8; ++r) {
          const int mg = m0 + wave * 16 + ((lane < 16) ? r : r + 8);
          if (ng > mg) s[nf][r] = -3.0e38f;
        }
      }
    }

    // ---- online softmax (row reductions within 16-lane halves) ----
    float nm[8], al[8];
#pragma unroll
    for (int r = 0; r < 8; ++r) {
      float mx = fmaxf(fmaxf(s[0][r], s[1][r]), fmaxf(s[2][r], s[3][r]));
#pragma unroll
      for (int d = 1; d < 16; d <<= 1) mx = fmaxf(mx, __shfl_xor(mx, d, 32));
      const float newm = fmaxf(rm[r], mx);
      al[r] = __expf(rm[r] - newm);
      nm[r] = newm; rm[r] = newm;
    }
#pragma unroll
    for (int nf = 0; nf < 4; ++nf)
#pragma unroll
      for (int r = 0; r < 8; ++r) s[nf][r] = __expf(s[nf][r] - nm[r]);
#pragma unroll
    for (int r = 0; r < 8; ++r) {
      float rs = s[0][r] + s[1][r] + s[2][r] + s[3][r];
#pragma unroll
      for (int d = 1; d < 16; d <<= 1) rs += __shfl_xor(rs, d, 32);
      rl[r] = rl[r] * al[r] + rs;
    }
#pragma unroll
    for (int nfo = 0; nfo < 8; ++nfo)
#pragma unroll
      for (int r = 0; r < 8; ++r) o[nfo][r] *= al[r];

    // ---- transpose P (C/D layout -> A layout) through per-wave LDS ----
    const int prow = (lane < 16) ? 0 : 8;
#pragma unroll
    for (int nf = 0; nf < 4; ++nf)
#pragma unroll
      for (int r = 0; r < 8; ++r)
        ps[wave][r + prow][nf * 16 + (lane & 15)] = (bf16)s[nf][r];
    asm volatile("s_wait_dscnt 0" ::: "memory");

    // ---- O += P * V  (K = 64 kv, N = 128 d) ----
#pragma unroll
    for (int k2 = 0; k2 < 2; ++k2) {
      const bf16x16 pa = ld_frag_a(&ps[wave][0][k2 * 32], 72);
#pragma unroll
      for (int nfo = 0; nfo < 8; ++nfo) {
        const bf16x16 bv = ld_frag_b(&vs[nfo * 16][k2 * 32], 72);
        o[nfo] = wmma_bf16(pa, bv, o[nfo]);
      }
    }
    __syncthreads();
  }

  // ---- epilogue: normalize and emit bf16 y rows (B,T,H*D) ----
#pragma unroll
  for (int r = 0; r < 8; ++r) {
    const float inv = 1.f / rl[r];
    const int mg = m0 + wave * 16 + ((lane < 16) ? r : r + 8);
    bf16* yrow = yb + ((size_t)b * TT + mg) * EE + h * DD;
#pragma unroll
    for (int nfo = 0; nfo < 8; ++nfo)
      yrow[nfo * 16 + (lane & 15)] = (bf16)(o[nfo][r] * inv);
  }
}

// ---------------------------------------------------------------------------
// Host-side orchestration
// ---------------------------------------------------------------------------
extern "C" void kernel_launch(void* const* d_in, const int* in_sizes, int n_in,
                              void* d_out, int out_size, void* d_ws, size_t ws_size,
                              hipStream_t stream) {
  const float* x    = (const float*)d_in[0];
  const float* cosb = (const float*)d_in[1];
  const float* sinb = (const float*)d_in[2];
  const float* WaQ  = (const float*)d_in[3];
  const float* WbQ  = (const float*)d_in[4];
  const float* WaK  = (const float*)d_in[5];
  const float* WbK  = (const float*)d_in[6];
  const float* WaV  = (const float*)d_in[7];
  const float* WbV  = (const float*)d_in[8];
  const float* Wc   = (const float*)d_in[9];
  float* out = (float*)d_out;

  char* ws = (char*)d_ws;
  size_t off = 0;
  auto alloc = [&](size_t bytes) {
    char* p = ws + off;
    off += (bytes + 255) & ~(size_t)255;
    return p;
  };
  bf16* xb    = (bf16*)alloc((size_t)BT * EE * 2);        // x in bf16
  bf16* wcat  = (bf16*)alloc((size_t)NPROJ * EE * 2);     // concat weights (NxK)
  bf16* wcb   = (bf16*)alloc((size_t)EE * EE * 2);        // W_cproj bf16 (NxK)
  float* proj = (float*)alloc((size_t)BT * NPROJ * 4);    // fused projections
  bf16* qb    = (bf16*)alloc((size_t)BB * HH * TT * DD * 2);
  bf16* kbf   = (bf16*)alloc((size_t)BB * HH * TT * DD * 2);
  bf16* vtb   = (bf16*)alloc((size_t)BB * HH * DD * TT * 2);
  bf16* yb    = (bf16*)alloc((size_t)BT * EE * 2);
  (void)ws_size; (void)in_sizes; (void)n_in; (void)out_size;

  // 1) precision conversion (bf16 operands, fp32 accumulate everywhere)
  f2bf_kernel<<<4096, 256, 0, stream>>>(x, xb, (long)BT * EE);
  f2bf_kernel<<<4096, 256, 0, stream>>>(Wc, wcb, (long)EE * EE);
  build_wcat_kernel<<<(NPROJ * EE) / 256, 256, 0, stream>>>(
      WaQ, WbQ, WaK, WbK, WaV, WbV, wcat);

  // 2) fused projection GEMM: proj[BT x 1536] = x * Wcat^T
  wmma_gemm_bf16<<<dim3(NPROJ / 64, BT / 128), 256, 0, stream>>>(
      xb, wcat, proj, EE, EE, EE, NPROJ);

  // 3) TPA combine + RoPE + rmsnorm -> Q, K, V^T
  tpa_combine_kernel<<<(BB * TT * HH) / 8, 256, 0, stream>>>(
      proj, cosb, sinb, qb, kbf, vtb);

  // 4) causal flash attention -> y (bf16, B,T,E)
  attn_kernel<<<dim3(TT / 64, HH, BB), 128, 0, stream>>>(qb, kbf, vtb, yb);

  // 5) output projection: out[BT x E] = y * Wc^T
  wmma_gemm_bf16<<<dim3(EE / 64, BT / 128), 256, 0, stream>>>(
      yb, wcb, out, EE, EE, EE, EE);
}
